// EGNNLayer_22402549416673
// MI455X (gfx1250) — matrix operand order repaired and checked by reading, not compiled, not run on hardware
//
#include <hip/hip_runtime.h>
#include <hip/hip_bf16.h>
#include <stdint.h>

#define HDIM  128
#define NNODE 20000
#define NEDGE 640000

typedef __attribute__((ext_vector_type(16))) _Float16 v16h;
typedef __attribute__((ext_vector_type(8)))  float    v8f;
typedef __attribute__((ext_vector_type(4)))  uint32_t u32x4;
typedef __attribute__((ext_vector_type(2)))  uint32_t u32x2;

union ABfrag { v16h h; uint32_t u[8]; u32x4 q[2]; };

__device__ __forceinline__ float silu_f(float x) {
  return x * (1.0f / (1.0f + __expf(-x)));
}

__device__ __forceinline__ void atomic_add_f32(float* p, float v) {
  unsafeAtomicAdd(p, v);
}

// Load one B fragment: 8 contiguous dwords per lane (2x b128).
__device__ __forceinline__ void load_bfrag(ABfrag& B, const uint32_t* __restrict__ Wf,
                                           int site, int lane) {
  const u32x4* p = (const u32x4*)(Wf + ((size_t)(site * 32 + lane)) * 8);
  B.q[0] = p[0];
  B.q[1] = p[1];
}

// Load one A fragment from LDS row (stride = K-dim of the activation buffer).
__device__ __forceinline__ void load_afrag(ABfrag& A, const _Float16* rowp,
                                           int k0, int hf) {
  A.q[0] = *(const u32x4*)(rowp + k0 + 16 * hf);
  A.q[1] = *(const u32x4*)(rowp + k0 + 16 * hf + 8);
}

// ---------------------------------------------------------------------------
// Repack f32 weights [K x 128] into fragment-major f16-pair layout:
//   dst[((t*KS + ks)*32 + lane)*8 + v] = {lo: W[2*pr][col], hi: W[2*pr+1][col]}
//   pr = ks*16 + 8*hf + v, col = t*16 + nl  (hf = lane>>4, nl = lane&15)
// so a lane's 8 B-fragment dwords are contiguous (-> 2x global_load_b128).
// ---------------------------------------------------------------------------
__global__ void pack_frag_kernel(const float* __restrict__ src,
                                 uint32_t* __restrict__ dst,
                                 int KS, int total) {
  int i = blockIdx.x * 256 + threadIdx.x;
  if (i >= total) return;
  int v    = i & 7;
  int lane = (i >> 3) & 31;
  int rest = i >> 8;
  int ks   = rest % KS;
  int t    = rest / KS;
  int nl = lane & 15, hf = lane >> 4;
  int pr  = ks * 16 + 8 * hf + v;
  int col = t * 16 + nl;
  union { _Float16 hh[2]; uint32_t u; } p;
  p.hh[0] = (_Float16)src[(2 * pr) * HDIM + col];
  p.hh[1] = (_Float16)src[(2 * pr + 1) * HDIM + col];
  dst[i] = p.u;
}

// ---------------------------------------------------------------------------
// Edge kernel: per wave = 16 edges.
//   L1: [16 x 257] @ eW1 (+eb1, +rank-1 sqd term) -> SiLU  -> a2 (f16)
//   L2: [16 x 128] @ eW2 (+eb2) -> SiLU -> m_ij  (atomicAdd to m_i, f16 -> a2)
//   L3: [16 x 128] @ cW1 (+cb1) -> SiLU -> dot cW2 -> coord weight -> pos atomics
// ---------------------------------------------------------------------------
__global__ __launch_bounds__(128) void egnn_edge_kernel(
    const float* __restrict__ hfeat, const float* __restrict__ pos,
    const int* __restrict__ ei,
    const float* __restrict__ eW1, const float* __restrict__ eb1,
    const float* __restrict__ eb2, const float* __restrict__ cb1,
    const float* __restrict__ cW2,
    const uint32_t* __restrict__ eW1f, const uint32_t* __restrict__ eW2f,
    const uint32_t* __restrict__ cW1f,
    float* __restrict__ m_i, float* __restrict__ pos_sum,
    float* __restrict__ cnt)
{
  __shared__ __align__(16) _Float16 s_ein[4][16 * 256];  // gathered edge inputs
  __shared__ __align__(16) _Float16 s_a2 [4][16 * 128];  // activations (f16)
  __shared__ float s_sqd[4][16];
  __shared__ float s_cw2[128];

  const int tid  = threadIdx.x;
  const int wave = tid >> 5;
  const int lane = tid & 31;
  const int nl   = lane & 15;
  const int hf   = lane >> 4;
  const int ebase = blockIdx.x * 64 + wave * 16;

  s_cw2[tid] = cW2[tid];

  if (lane < 16) {
    int e = ebase + lane;
    int r = ei[e], c = ei[NEDGE + e];
    float dx = pos[r * 3 + 0] - pos[c * 3 + 0];
    float dy = pos[r * 3 + 1] - pos[c * 3 + 1];
    float dz = pos[r * 3 + 2] - pos[c * 3 + 2];
    s_sqd[wave][lane] = dx * dx + dy * dy + dz * dz;
  }

  _Float16* einw = s_ein[wave];
  _Float16* a2w  = s_a2[wave];

  // Gather [h[row] || h[col]] -> f16 LDS: one float4 load + one b64 LDS store.
  #pragma unroll 2
  for (int it = 0; it < 32; ++it) {
    int e   = it >> 1;
    int sel = it & 1;
    int node = ei[sel * NEDGE + ebase + e];
    const float4 x = *(const float4*)(hfeat + node * HDIM + lane * 4);
    union { _Float16 hh[4]; u32x2 d; } pk;
    pk.hh[0] = (_Float16)x.x; pk.hh[1] = (_Float16)x.y;
    pk.hh[2] = (_Float16)x.z; pk.hh[3] = (_Float16)x.w;
    *(u32x2*)(einw + e * 256 + sel * 128 + lane * 4) = pk.d;
  }
  __syncthreads();

  int rowv[8];
  #pragma unroll
  for (int v = 0; v < 8; ++v) rowv[v] = ei[ebase + v + 8 * hf];

  // ---- Layer 1 (K = 256 via WMMA + rank-1 sq_dists term) ----
  #pragma unroll 1
  for (int t = 0; t < 8; ++t) {
    const int n0 = t * 16;
    float b = eb1[n0 + nl];
    v8f acc;
    #pragma unroll
    for (int v = 0; v < 8; ++v) acc[v] = b;
    #pragma unroll
    for (int ks = 0; ks < 8; ++ks) {
      ABfrag A, B;
      load_afrag(A, einw + nl * 256, ks * 32, hf);
      load_bfrag(B, eW1f, t * 8 + ks, lane);
      acc = __builtin_amdgcn_wmma_f32_16x16x32_f16(false, A.h, false, B.h,
                                                   (short)0, acc, false, false);
    }
    float wlast = eW1[256 * HDIM + n0 + nl];  // 257th input row: sq_dists
    #pragma unroll
    for (int v = 0; v < 8; ++v) {
      float x = acc[v] + s_sqd[wave][v + 8 * hf] * wlast;
      a2w[(v + 8 * hf) * HDIM + n0 + nl] = (_Float16)silu_f(x);
    }
  }
  __syncthreads();

  // ---- Layer 2 (m_ij): keep f32 results in regs, scatter-add to m_i ----
  v8f mreg[8];
  #pragma unroll
  for (int t = 0; t < 8; ++t) {
    const int n0 = t * 16;
    float b = eb2[n0 + nl];
    v8f acc;
    #pragma unroll
    for (int v = 0; v < 8; ++v) acc[v] = b;
    #pragma unroll
    for (int ks = 0; ks < 4; ++ks) {
      ABfrag A, B;
      load_afrag(A, a2w + nl * HDIM, ks * 32, hf);
      load_bfrag(B, eW2f, t * 4 + ks, lane);
      acc = __builtin_amdgcn_wmma_f32_16x16x32_f16(false, A.h, false, B.h,
                                                   (short)0, acc, false, false);
    }
    #pragma unroll
    for (int v = 0; v < 8; ++v) {
      float x = silu_f(acc[v]);
      mreg[t][v] = x;
      atomic_add_f32(&m_i[rowv[v] * HDIM + n0 + nl], x);
    }
  }
  // overwrite a2 with m_ij (f16) for the coord MLP
  #pragma unroll
  for (int t = 0; t < 8; ++t) {
    #pragma unroll
    for (int v = 0; v < 8; ++v)
      a2w[(v + 8 * hf) * HDIM + t * 16 + nl] = (_Float16)mreg[t][v];
  }
  __syncthreads();

  // ---- Layer 3 (coord MLP hidden) + dot with cW2 ----
  v8f pacc;
  #pragma unroll
  for (int v = 0; v < 8; ++v) pacc[v] = 0.0f;
  #pragma unroll 1
  for (int t = 0; t < 8; ++t) {
    const int n0 = t * 16;
    float b = cb1[n0 + nl];
    v8f acc;
    #pragma unroll
    for (int v = 0; v < 8; ++v) acc[v] = b;
    #pragma unroll
    for (int ks = 0; ks < 4; ++ks) {
      ABfrag A, B;
      load_afrag(A, a2w + nl * HDIM, ks * 32, hf);
      load_bfrag(B, cW1f, t * 4 + ks, lane);
      acc = __builtin_amdgcn_wmma_f32_16x16x32_f16(false, A.h, false, B.h,
                                                   (short)0, acc, false, false);
    }
    float cw = s_cw2[n0 + nl];
    #pragma unroll
    for (int v = 0; v < 8; ++v) pacc[v] += silu_f(acc[v]) * cw;
  }
  // reduce partial dots across the 16 column lanes (stays within lane half)
  #pragma unroll
  for (int off = 8; off >= 1; off >>= 1) {
    #pragma unroll
    for (int v = 0; v < 8; ++v) pacc[v] += __shfl_xor(pacc[v], off, 32);
  }

  if (nl == 0) {
    #pragma unroll
    for (int v = 0; v < 8; ++v) {
      int e = ebase + v + 8 * hf;
      int r = rowv[v];
      int c = ei[NEDGE + e];
      float s = s_sqd[wave][v + 8 * hf];
      float scale = rsqrtf(s + 1e-8f) * pacc[v];
      atomic_add_f32(&pos_sum[r * 3 + 0], (pos[r * 3 + 0] - pos[c * 3 + 0]) * scale);
      atomic_add_f32(&pos_sum[r * 3 + 1], (pos[r * 3 + 1] - pos[c * 3 + 1]) * scale);
      atomic_add_f32(&pos_sum[r * 3 + 2], (pos[r * 3 + 2] - pos[c * 3 + 2]) * scale);
      atomic_add_f32(&cnt[r], 1.0f);
    }
  }
}

// ---------------------------------------------------------------------------
// Node kernel: h_out = h + nMLP(concat(h, m_i)); wave = 16 nodes.
// ---------------------------------------------------------------------------
__global__ __launch_bounds__(128) void egnn_node_kernel(
    const float* __restrict__ hfeat, const float* __restrict__ m_i,
    const float* __restrict__ nb1, const float* __restrict__ nb2,
    const uint32_t* __restrict__ nW1f, const uint32_t* __restrict__ nW2f,
    float* __restrict__ hout)
{
  __shared__ __align__(16) _Float16 s_ein[4][16 * 256];
  __shared__ __align__(16) _Float16 s_a2 [4][16 * 128];

  const int tid  = threadIdx.x;
  const int wave = tid >> 5;
  const int lane = tid & 31;
  const int nl   = lane & 15;
  const int hf   = lane >> 4;
  int base = blockIdx.x * 64 + wave * 16;
  const bool valid = (base < NNODE);
  if (!valid) base = 0;  // clamp; still participate in barriers

  _Float16* einw = s_ein[wave];
  _Float16* a2w  = s_a2[wave];

  #pragma unroll 2
  for (int it = 0; it < 32; ++it) {
    int i   = it >> 1;
    int sel = it & 1;
    const float* srcp = sel ? (m_i + (size_t)(base + i) * HDIM)
                            : (hfeat + (size_t)(base + i) * HDIM);
    const float4 x = *(const float4*)(srcp + lane * 4);
    union { _Float16 hh[4]; u32x2 d; } pk;
    pk.hh[0] = (_Float16)x.x; pk.hh[1] = (_Float16)x.y;
    pk.hh[2] = (_Float16)x.z; pk.hh[3] = (_Float16)x.w;
    *(u32x2*)(einw + i * 256 + sel * 128 + lane * 4) = pk.d;
  }
  __syncthreads();

  #pragma unroll 1
  for (int t = 0; t < 8; ++t) {
    const int n0 = t * 16;
    float b = nb1[n0 + nl];
    v8f acc;
    #pragma unroll
    for (int v = 0; v < 8; ++v) acc[v] = b;
    #pragma unroll
    for (int ks = 0; ks < 8; ++ks) {
      ABfrag A, B;
      load_afrag(A, einw + nl * 256, ks * 32, hf);
      load_bfrag(B, nW1f, t * 8 + ks, lane);
      acc = __builtin_amdgcn_wmma_f32_16x16x32_f16(false, A.h, false, B.h,
                                                   (short)0, acc, false, false);
    }
    #pragma unroll
    for (int v = 0; v < 8; ++v)
      a2w[(v + 8 * hf) * HDIM + n0 + nl] = (_Float16)silu_f(acc[v]);
  }
  __syncthreads();

  #pragma unroll 1
  for (int t = 0; t < 8; ++t) {
    const int n0 = t * 16;
    float b = nb2[n0 + nl];
    v8f acc;
    #pragma unroll
    for (int v = 0; v < 8; ++v) acc[v] = b;
    #pragma unroll
    for (int ks = 0; ks < 4; ++ks) {
      ABfrag A, B;
      load_afrag(A, a2w + nl * HDIM, ks * 32, hf);
      load_bfrag(B, nW2f, t * 4 + ks, lane);
      acc = __builtin_amdgcn_wmma_f32_16x16x32_f16(false, A.h, false, B.h,
                                                   (short)0, acc, false, false);
    }
    if (valid) {
      #pragma unroll
      for (int v = 0; v < 8; ++v) {
        int node = base + v + 8 * hf;
        hout[node * HDIM + n0 + nl] = acc[v] + hfeat[node * HDIM + n0 + nl];
      }
    }
  }
}

__global__ void pos_finalize_kernel(const float* __restrict__ pos,
                                    const float* __restrict__ pos_sum,
                                    const float* __restrict__ cnt,
                                    float* __restrict__ pout) {
  int i = blockIdx.x * 256 + threadIdx.x;
  if (i >= NNODE * 3) return;
  int nidx = i / 3;
  float c = fmaxf(cnt[nidx], 1.0f);
  pout[i] = pos[i] + pos_sum[i] / c;
}

extern "C" void kernel_launch(void* const* d_in, const int* in_sizes, int n_in,
                              void* d_out, int out_size, void* d_ws, size_t ws_size,
                              hipStream_t stream) {
  (void)in_sizes; (void)n_in; (void)out_size; (void)ws_size;
  const float* hfeat = (const float*)d_in[0];
  const float* pos   = (const float*)d_in[1];
  const int*   ei    = (const int*)d_in[2];
  const float* eW1   = (const float*)d_in[3];
  const float* eb1   = (const float*)d_in[4];
  const float* eW2   = (const float*)d_in[5];
  const float* eb2   = (const float*)d_in[6];
  const float* cW1   = (const float*)d_in[7];
  const float* cb1   = (const float*)d_in[8];
  const float* cW2   = (const float*)d_in[9];
  const float* nW1   = (const float*)d_in[10];
  const float* nb1   = (const float*)d_in[11];
  const float* nW2   = (const float*)d_in[12];
  const float* nb2   = (const float*)d_in[13];

  char* ws = (char*)d_ws;
  float*    m_i     = (float*)(ws + 0);            // N*128 f32
  float*    pos_sum = (float*)(ws + 10240000);     // N*3  f32
  float*    cnt     = (float*)(ws + 10480000);     // N    f32
  uint32_t* eW1f    = (uint32_t*)(ws + 10560000);  // 16384 u32 (frag-major)
  uint32_t* eW2f    = (uint32_t*)(ws + 10625536);  // 8192
  uint32_t* cW1f    = (uint32_t*)(ws + 10658304);  // 8192
  uint32_t* nW1f    = (uint32_t*)(ws + 10691072);  // 16384
  uint32_t* nW2f    = (uint32_t*)(ws + 10756608);  // 8192

  // zero the scatter accumulators every call (ws is poisoned once, not restored)
  hipMemsetAsync(m_i, 0, (size_t)(NNODE * 128 + NNODE * 3 + NNODE) * sizeof(float), stream);

  pack_frag_kernel<<<(16384 + 255) / 256, 256, 0, stream>>>(eW1, eW1f, 8, 16384);
  pack_frag_kernel<<<(8192 + 255) / 256, 256, 0, stream>>>(eW2, eW2f, 4, 8192);
  pack_frag_kernel<<<(8192 + 255) / 256, 256, 0, stream>>>(cW1, cW1f, 4, 8192);
  pack_frag_kernel<<<(16384 + 255) / 256, 256, 0, stream>>>(nW1, nW1f, 8, 16384);
  pack_frag_kernel<<<(8192 + 255) / 256, 256, 0, stream>>>(nW2, nW2f, 4, 8192);

  egnn_edge_kernel<<<NEDGE / 64, 128, 0, stream>>>(
      hfeat, pos, ei, eW1, eb1, eb2, cb1, cW2, eW1f, eW2f, cW1f,
      m_i, pos_sum, cnt);

  float* hout = (float*)d_out;
  float* pout = (float*)d_out + (size_t)NNODE * HDIM;
  egnn_node_kernel<<<(NNODE + 63) / 64, 128, 0, stream>>>(
      hfeat, m_i, nb1, nb2, nW1f, nW2f, hout);
  pos_finalize_kernel<<<(NNODE * 3 + 255) / 256, 256, 0, stream>>>(
      pos, pos_sum, cnt, pout);
}
